// GraphConvLayer_75316546503241
// MI455X (gfx1250) — matrix operand order repaired and checked by reading, not compiled, not run on hardware
//
#include <hip/hip_runtime.h>

// ---------------------------------------------------------------------------
// GraphConv on MI455X (gfx1250, wave32).
//   out = x@W_lin + b_lin + segment_sum_row( (w_e * x[col]) @ W_agg + b_agg )
// Linearity trick: scatter first (h[i] = sum w_e * x[col_e], deg[i] = #edges),
// then  out = x@W_lin + h@W_agg + b_lin + deg*b_agg     (16x fewer GEMM FLOPs).
// x (51MB) and h (51MB) both fit in the 192MB L2 -> edge gathers/atomics are
// on-chip. GEMM uses v_wmma_f32_16x16x32_bf16 with a bf16 hi/lo split
// (hi*hi + hi*lo + lo*hi) for ~fp32 accuracy; workload is bandwidth-bound so
// the 3x matrix-op count is free.
// Round-2 changes: weights pre-split/pre-packed once into fragment order
// (removes per-wave conversion VALU + scalar column loads), and the x/h node
// tile is split once per block into padded bf16 LDS arrays (ds_load_b128
// fragment fetches, 8x less conversion work).
// ---------------------------------------------------------------------------

typedef __attribute__((ext_vector_type(16))) __bf16 v16bf;
typedef __attribute__((ext_vector_type(8)))  __bf16 v8bf;
typedef __attribute__((ext_vector_type(8)))  float  v8f;

__device__ __forceinline__ __bf16 f2bf(float f) {
  unsigned u = __builtin_bit_cast(unsigned, f);
  u += 0x7FFFu + ((u >> 16) & 1u);            // round-to-nearest-even
  unsigned short s = (unsigned short)(u >> 16);
  return __builtin_bit_cast(__bf16, s);
}
__device__ __forceinline__ float bf2f(__bf16 b) {
  unsigned short s = __builtin_bit_cast(unsigned short, b);
  unsigned u = ((unsigned)s) << 16;
  return __builtin_bit_cast(float, u);
}
__device__ __forceinline__ void split_bf16(float f, __bf16& hi, __bf16& lo) {
  hi = f2bf(f);
  lo = f2bf(f - bf2f(hi));
}

// ---------------- kernel 0: zero workspace (h accumulator + deg) -----------
__global__ __launch_bounds__(256) void gc_zero_f32(float* __restrict__ p, int n) {
  int i = blockIdx.x * blockDim.x + threadIdx.x;
  if (i < n) p[i] = 0.0f;
}

// ---------------- kernel 1: pre-split weights into B-fragment order --------
// Packed layout per matrix/part: [kk(8)][n(128)][e(16)] bf16, where the
// source element is W[(kk*16+e)*128 + n]. A wave's B fragment for k-block
// kb and column n is then 16 consecutive bf16 = one aligned 32B load.
__global__ __launch_bounds__(256) void gc_pack_weights(
    const float* __restrict__ Wl, const float* __restrict__ Wa,
    __bf16* __restrict__ wl_hi, __bf16* __restrict__ wl_lo,
    __bf16* __restrict__ wa_hi, __bf16* __restrict__ wa_lo) {
  int idx = blockIdx.x * blockDim.x + threadIdx.x;   // over 128*128 elements
  if (idx >= 128 * 128) return;
  int k = idx >> 7;
  int n = idx & 127;
  int kk = k >> 4;
  int e  = k & 15;
  int o  = ((kk << 7) + n) * 16 + e;
  __bf16 hi, lo;
  split_bf16(Wl[idx], hi, lo);
  wl_hi[o] = hi; wl_lo[o] = lo;
  split_bf16(Wa[idx], hi, lo);
  wa_hi[o] = hi; wa_lo[o] = lo;
}

// ---------------- kernel 2: weighted neighbor scatter ----------------------
// One wave32 per edge; lane l handles 4 contiguous features (32*4 = 128).
__global__ __launch_bounds__(256) void gc_edge_scatter(
    const float* __restrict__ x,
    const long long* __restrict__ row,   // edge_index[0,:]
    const long long* __restrict__ col,   // edge_index[1,:]
    const float* __restrict__ ew,
    float* __restrict__ h,
    float* __restrict__ deg,
    int nE) {
  int gid  = blockIdx.x * blockDim.x + threadIdx.x;
  int e    = gid >> 5;
  int lane = gid & 31;
  if (e >= nE) return;
  long long r = row[e];
  long long c = col[e];
  float     w = ew[e];
  float4 v = ((const float4*)(x + (size_t)c * 128))[lane];
  float* dst = h + (size_t)r * 128 + lane * 4;
  atomicAdd(dst + 0, v.x * w);
  atomicAdd(dst + 1, v.y * w);
  atomicAdd(dst + 2, v.z * w);
  atomicAdd(dst + 3, v.w * w);
  if (lane == 0) atomicAdd(deg + r, 1.0f);
}

// ---------------- kernel 3: fused dual-GEMM + bias epilogue ----------------
// Block = 256 threads = 8 waves. Block owns a 16-row node tile; wave w owns
// output columns [16w, 16w+16). x/h tiles are split to bf16 hi/lo in LDS once
// per block; A fragments come from ds_load_b128, B fragments from the packed
// weight arrays. 24 chained v_wmma_f32_16x16x32_bf16 per wave.
#define LDS_STRIDE 136   // 128 + 8 bf16 pad (272B rows) -> no bank conflicts

__global__ __launch_bounds__(256) void gc_fused_gemm(
    const float* __restrict__ x,
    const float* __restrict__ hagg,
    const __bf16* __restrict__ wl_hi, const __bf16* __restrict__ wl_lo,
    const __bf16* __restrict__ wa_hi, const __bf16* __restrict__ wa_lo,
    const float* __restrict__ bl, const float* __restrict__ ba,
    const float* __restrict__ deg,
    float* __restrict__ out, int N) {
  const int D = 128;
  __shared__ __bf16 sx_hi[16 * LDS_STRIDE];
  __shared__ __bf16 sx_lo[16 * LDS_STRIDE];
  __shared__ __bf16 sh_hi[16 * LDS_STRIDE];
  __shared__ __bf16 sh_lo[16 * LDS_STRIDE];

  int lane  = threadIdx.x & 31;
  int warp  = threadIdx.x >> 5;
  int half  = lane >> 4;            // which 16-lane half of the wave
  int l16   = lane & 15;
  int tileM = blockIdx.x * 16;
  int tileN = warp * 16;

  // ---- cooperative load + split of the 16x128 x/h tiles into LDS
  for (int idx = threadIdx.x; idx < 16 * D; idx += 256) {
    int r = idx >> 7;
    int k = idx & 127;
    int m = tileM + r;
    if (m >= N) m = N - 1;          // clamp: keeps data valid for full EXEC
    __bf16 hi, lo;
    split_bf16(x[(size_t)m * D + k], hi, lo);
    sx_hi[r * LDS_STRIDE + k] = hi;
    sx_lo[r * LDS_STRIDE + k] = lo;
    split_bf16(hagg[(size_t)m * D + k], hi, lo);
    sh_hi[r * LDS_STRIDE + k] = hi;
    sh_lo[r * LDS_STRIDE + k] = lo;
  }
  __syncthreads();

  int ncol = tileN + l16;
  int koff = half * 16;              // A-frag K offset per lane half
  int abase = l16 * LDS_STRIDE + koff;

  v8f c = {};

#pragma unroll
  for (int kb = 0; kb < D; kb += 32) {
    // ---- A fragments (16x32 bf16): elem e -> K=kb+koff+e (koff = 16*half)
    const __bf16* pxh = sx_hi + abase + kb;
    const __bf16* pxl = sx_lo + abase + kb;
    const __bf16* phh = sh_hi + abase + kb;
    const __bf16* phl = sh_lo + abase + kb;
    v8bf x0 = *(const v8bf*)(pxh);      v8bf x1 = *(const v8bf*)(pxh + 8);
    v8bf y0 = *(const v8bf*)(pxl);      v8bf y1 = *(const v8bf*)(pxl + 8);
    v8bf g0 = *(const v8bf*)(phh);      v8bf g1 = *(const v8bf*)(phh + 8);
    v8bf f0 = *(const v8bf*)(phl);      v8bf f1 = *(const v8bf*)(phl + 8);
    v16bf ax_hi = __builtin_shufflevector(x0, x1, 0,1,2,3,4,5,6,7,8,9,10,11,12,13,14,15);
    v16bf ax_lo = __builtin_shufflevector(y0, y1, 0,1,2,3,4,5,6,7,8,9,10,11,12,13,14,15);
    v16bf ah_hi = __builtin_shufflevector(g0, g1, 0,1,2,3,4,5,6,7,8,9,10,11,12,13,14,15);
    v16bf ah_lo = __builtin_shufflevector(f0, f1, 0,1,2,3,4,5,6,7,8,9,10,11,12,13,14,15);

    // ---- B fragments: packed, elem e -> K = kb + 16*half + e, N = ncol
    int kk = (kb >> 4) + half;
    size_t bo = ((size_t)((kk << 7) + ncol)) * 16;
    v16bf bwl_hi = *(const v16bf*)(wl_hi + bo);
    v16bf bwl_lo = *(const v16bf*)(wl_lo + bo);
    v16bf bwa_hi = *(const v16bf*)(wa_hi + bo);
    v16bf bwa_lo = *(const v16bf*)(wa_lo + bo);

    // ---- 6 WMMAs, all chained into one f32 accumulator
    c = __builtin_amdgcn_wmma_f32_16x16x32_bf16(false, ax_hi, false, bwl_hi, (short)0, c, false, false);
    c = __builtin_amdgcn_wmma_f32_16x16x32_bf16(false, ax_hi, false, bwl_lo, (short)0, c, false, false);
    c = __builtin_amdgcn_wmma_f32_16x16x32_bf16(false, ax_lo, false, bwl_hi, (short)0, c, false, false);
    c = __builtin_amdgcn_wmma_f32_16x16x32_bf16(false, ah_hi, false, bwa_hi, (short)0, c, false, false);
    c = __builtin_amdgcn_wmma_f32_16x16x32_bf16(false, ah_hi, false, bwa_lo, (short)0, c, false, false);
    c = __builtin_amdgcn_wmma_f32_16x16x32_bf16(false, ah_lo, false, bwa_hi, (short)0, c, false, false);
  }

  // ---- epilogue: C layout VGPR r -> M = tileM + r + 8*half, N = ncol
  float blv = bl[ncol];
  float bav = ba[ncol];
#pragma unroll
  for (int r = 0; r < 8; ++r) {
    int m = tileM + r + half * 8;
    if (m < N) {
      float d = deg[m];
      out[(size_t)m * D + ncol] = c[r] + blv + d * bav;
    }
  }
}

// ---------------------------------------------------------------------------
extern "C" void kernel_launch(void* const* d_in, const int* in_sizes, int n_in,
                              void* d_out, int out_size, void* d_ws, size_t ws_size,
                              hipStream_t stream) {
  const float*     x  = (const float*)d_in[0];
  const long long* ei = (const long long*)d_in[1];   // [2, E] int64
  const float*     ew = (const float*)d_in[2];
  const float*     Wl = (const float*)d_in[3];
  const float*     bl = (const float*)d_in[4];
  const float*     Wa = (const float*)d_in[5];
  const float*     ba = (const float*)d_in[6];
  float*           out = (float*)d_out;

  const int D = 128;
  const int N = in_sizes[0] / D;     // 100000
  const int E = in_sizes[2];         // 1600000

  // workspace layout
  float*  h    = (float*)d_ws;                        // [N,128] f32
  float*  deg  = h + (size_t)N * D;                   // [N]     f32
  __bf16* wp   = (__bf16*)(deg + N);                  // packed weights
  const int WSZ = 128 * 128;                          // 16384 elems each
  __bf16* wl_hi = wp;
  __bf16* wl_lo = wp + WSZ;
  __bf16* wa_hi = wp + 2 * WSZ;
  __bf16* wa_lo = wp + 3 * WSZ;

  // 1) zero accumulators (workspace is poisoned; must re-zero every call)
  int zn = N * D + N;
  gc_zero_f32<<<(zn + 255) / 256, 256, 0, stream>>>(h, zn);

  // 2) pre-split weights into fragment-ordered bf16 hi/lo (tiny)
  gc_pack_weights<<<(WSZ + 255) / 256, 256, 0, stream>>>(Wl, Wa, wl_hi, wl_lo, wa_hi, wa_lo);

  // 3) weighted scatter: one wave32 per edge, atomics land in L2
  long long tE = (long long)E * 32;
  int blocksE = (int)((tE + 255) / 256);
  gc_edge_scatter<<<blocksE, 256, 0, stream>>>(x, ei, ei + E, ew, h, deg, E);

  // 4) fused dual GEMM (WMMA bf16x3, LDS-staged A) + bias epilogue
  int blocksG = (N + 15) / 16;
  gc_fused_gemm<<<blocksG, 256, 0, stream>>>(x, h, wl_hi, wl_lo, wa_hi, wa_lo,
                                             bl, ba, deg, out, N);
}
